// Model_55611236549533
// MI455X (gfx1250) — matrix operand order repaired and checked by reading, not compiled, not run on hardware
//
#include <hip/hip_runtime.h>
#include <hip/hip_bf16.h>

// ---------------------------------------------------------------------------
// Problem constants (from reference): N=1.5e6 rows, F=64, B=256, D=64,
// S = B*D = 16384 segments. MLP dims: 64 -> 128 -> 32 (x2 stacks).
// ---------------------------------------------------------------------------
#define KN    1500000
#define KF    64
#define KH1   128
#define KH2   32
#define KSEG  16384

// Packed weight fragments per stack: layer1 -> [k0<2][n0<8], layer2 -> [k0<4][n0<2]
// 24 fragments, each 32 lanes x 16 halves (1 KB). 24 KB per stack, 48 KB total.
#define NFRAG 24

typedef _Float16 v16h __attribute__((ext_vector_type(16)));
typedef _Float16 v8h  __attribute__((ext_vector_type(8)));
typedef float    v8f  __attribute__((ext_vector_type(8)));

// ---------------------------------------------------------------------------
// WMMA helper: D = A(16x32 f16) * B(32x16 f16) + C(16x16 f32)
// ---------------------------------------------------------------------------
__device__ __forceinline__ v8f wmma_f16(v16h a, v16h b, v8f c) {
  return __builtin_amdgcn_wmma_f32_16x16x32_f16(
      /*neg_a=*/false, a, /*neg_b=*/false, b,
      /*c_mod=*/(short)0, c, /*reuse_a=*/false, /*reuse_b=*/false);
}

__device__ __forceinline__ void cvt4(v16h& a, int o, float4 u) {
  a[o + 0] = (_Float16)u.x;
  a[o + 1] = (_Float16)u.y;
  a[o + 2] = (_Float16)u.z;
  a[o + 3] = (_Float16)u.w;
}

// A-fragment (16x32 f16, M x K) from global f32 matrix row `base = X + row*64`.
// ISA layout: lane = hi*16 + m; halves[0..7] = A[m][k0*32 + hi*8 + j],
//             halves[8..15] = A[m][k0*32 + 16 + hi*8 + j].
__device__ __forceinline__ v16h load_a_global(const float* base, int k0, int hi) {
  const float4* q = reinterpret_cast<const float4*>(base + k0 * 32 + hi * 8);
  v16h a;
  cvt4(a, 0,  q[0]);
  cvt4(a, 4,  q[1]);
  cvt4(a, 8,  q[4]);   // +16 floats
  cvt4(a, 12, q[5]);
  return a;
}

// A-fragment from LDS f16 matrix with row stride 128 (layer-2 input).
__device__ __forceinline__ v16h load_a_lds(const _Float16* rowp, int k0, int hi) {
  const v8h* q = reinterpret_cast<const v8h*>(rowp + k0 * 32 + hi * 8);
  v8h lo = q[0];
  v8h up = q[2];       // +16 halves
  v16h a;
#pragma unroll
  for (int j = 0; j < 8; ++j) { a[j] = lo[j]; a[8 + j] = up[j]; }
  return a;
}

// Pre-packed B fragment: one 32B per-lane vector load (no cvt, no gather).
__device__ __forceinline__ v16h load_b_packed(const _Float16* base, int fragIdx,
                                              int lane) {
  return *reinterpret_cast<const v16h*>(base + fragIdx * 512 + lane * 16);
}

// ---------------------------------------------------------------------------
// Pre-kernel: convert f32 weights to WMMA-ready f16 B-fragments in workspace.
// ISA B layout (32x16 K x N): lane = hi*16 + n; halves[j] = B[k0*32+hi*8+(j<8?j:8+j)][col].
// ---------------------------------------------------------------------------
__global__ void __launch_bounds__(256)
pack_weight_frags(const float* __restrict__ tfc0_w, const float* __restrict__ tfc1_w,
                  const float* __restrict__ rl0_w,  const float* __restrict__ rl1_w,
                  _Float16* __restrict__ frags)
{
  const int s = blockIdx.x;                         // 0 = tfc stack, 1 = rl stack
  const float* W0 = s ? rl0_w : tfc0_w;             // [64 x 128]
  const float* W1 = s ? rl1_w : tfc1_w;             // [128 x 32]
  _Float16* out = frags + (size_t)s * NFRAG * 512;

  for (int item = threadIdx.x; item < NFRAG * 32; item += blockDim.x) {
    const int f    = item >> 5;
    const int lane = item & 31;
    const int hi   = lane >> 4;
    const int ln   = lane & 15;
    v16h h;
    if (f < 16) {                                   // layer 1: [k0<2][n0<8]
      const int k0 = f >> 3, n0 = f & 7;
      const int col = n0 * 16 + ln;
      const int kr  = k0 * 32 + hi * 8;
#pragma unroll
      for (int j = 0; j < 16; ++j) h[j] = (_Float16)W0[(kr + (j < 8 ? j : 8 + j)) * KH1 + col];
    } else {                                        // layer 2: [k0<4][n0<2]
      const int g = f - 16;
      const int k0 = g >> 1, n0 = g & 1;
      const int col = n0 * 16 + ln;
      const int kr  = k0 * 32 + hi * 8;
#pragma unroll
      for (int j = 0; j < 16; ++j) h[j] = (_Float16)W1[(kr + (j < 8 ? j : 8 + j)) * KH2 + col];
    }
    *reinterpret_cast<v16h*>(out + item * 16) = h;
  }
}

// ---------------------------------------------------------------------------
// One stack: h = relu(x@W0+B0) [rowsx128]; h = relu(h@W1+B1) [rowsx32];
// segment-sum h into seg[KSEG][32] via sorted-run atomics.
// Each wave owns 32 rows (2 M-tiles). No cross-wave LDS deps.
// ---------------------------------------------------------------------------
__device__ __forceinline__ void run_stack(
    const float* __restrict__ x,
    const _Float16* __restrict__ frag,              // packed B frags, this stack
    const float* __restrict__ B0, const float* __restrict__ B1,
    float* __restrict__ seg,
    _Float16* sH1, float* sH2, const int* sSID,
    int blockRow0, int wave, int lane)
{
  const int hi = lane >> 4;
  const int ln = lane & 15;
  const int wr0 = wave * 32;                 // local row base of this wave
  const int grow0 = blockRow0 + wr0;         // global row base

  // ---------------- layer 1: [32x64] @ [64x128] ----------------
  v8f acc[2][8];
#pragma unroll
  for (int mt = 0; mt < 2; ++mt)
#pragma unroll
    for (int n = 0; n < 8; ++n)
#pragma unroll
      for (int r = 0; r < 8; ++r) acc[mt][n][r] = 0.0f;

#pragma unroll
  for (int k0 = 0; k0 < 2; ++k0) {
    int r0 = grow0 + ln;       if (r0 >= KN) r0 = KN - 1;
    int r1 = grow0 + 16 + ln;  if (r1 >= KN) r1 = KN - 1;
    v16h a0 = load_a_global(x + (size_t)r0 * KF, k0, hi);
    v16h a1 = load_a_global(x + (size_t)r1 * KF, k0, hi);
#pragma unroll
    for (int n0 = 0; n0 < 8; ++n0) {
      v16h b = load_b_packed(frag, k0 * 8 + n0, lane);
      acc[0][n0] = wmma_f16(a0, b, acc[0][n0]);
      acc[1][n0] = wmma_f16(a1, b, acc[1][n0]);
    }
  }

  // bias + relu + f16 store to LDS (D layout: reg r -> row mt*16 + hi*8 + r)
#pragma unroll
  for (int n0 = 0; n0 < 8; ++n0) {
    int col = n0 * 16 + ln;
    float bias = B0[col];
#pragma unroll
    for (int mt = 0; mt < 2; ++mt)
#pragma unroll
      for (int r = 0; r < 8; ++r) {
        float v = acc[mt][n0][r] + bias;
        v = fmaxf(v, 0.0f);
        int lrow = wr0 + mt * 16 + hi * 8 + r;
        sH1[lrow * KH1 + col] = (_Float16)v;
      }
  }

  // ---------------- layer 2: [32x128] @ [128x32] ----------------
  v8f acc2[2][2];
#pragma unroll
  for (int mt = 0; mt < 2; ++mt)
#pragma unroll
    for (int n = 0; n < 2; ++n)
#pragma unroll
      for (int r = 0; r < 8; ++r) acc2[mt][n][r] = 0.0f;

#pragma unroll
  for (int k0 = 0; k0 < 4; ++k0) {
    v16h a0 = load_a_lds(sH1 + (wr0 + ln) * KH1, k0, hi);
    v16h a1 = load_a_lds(sH1 + (wr0 + 16 + ln) * KH1, k0, hi);
#pragma unroll
    for (int n0 = 0; n0 < 2; ++n0) {
      v16h b = load_b_packed(frag, 16 + k0 * 2 + n0, lane);
      acc2[0][n0] = wmma_f16(a0, b, acc2[0][n0]);
      acc2[1][n0] = wmma_f16(a1, b, acc2[1][n0]);
    }
  }

#pragma unroll
  for (int n0 = 0; n0 < 2; ++n0) {
    int col = n0 * 16 + ln;
    float bias = B1[col];
#pragma unroll
    for (int mt = 0; mt < 2; ++mt)
#pragma unroll
      for (int r = 0; r < 8; ++r) {
        float v = acc2[mt][n0][r] + bias;
        v = fmaxf(v, 0.0f);
        int lrow = wr0 + mt * 16 + hi * 8 + r;
        sH2[lrow * KH2 + col] = v;
      }
  }

  // ------------- sorted-run segment reduction (lane = channel) -------------
  float a = 0.0f;
  int cur = -1;
  for (int i = 0; i < 32; ++i) {
    int gr = blockRow0 + wr0 + i;
    if (gr >= KN) break;                     // rows are contiguous ascending
    int s = sSID[wr0 + i];
    if (s != cur) {
      if (cur >= 0) atomicAdd(&seg[(size_t)cur * 32 + lane], a);
      a = 0.0f;
      cur = s;
    }
    a += sH2[(wr0 + i) * KH2 + lane];
  }
  if (cur >= 0) atomicAdd(&seg[(size_t)cur * 32 + lane], a);
}

// ---------------------------------------------------------------------------
// Kernel 1: fused per-row MLPs (both stacks) + segment sums.
// 128 rows per block, 4 waves (wave32), WMMA f16 with f32 accumulate.
// ---------------------------------------------------------------------------
__global__ void __launch_bounds__(128)
fused_rowmlp_segsum(const float* __restrict__ x, const int* __restrict__ sid,
                    const _Float16* __restrict__ frags,
                    const float* __restrict__ tfc0_b, const float* __restrict__ tfc1_b,
                    const float* __restrict__ rl0_b,  const float* __restrict__ rl1_b,
                    float* __restrict__ segA, float* __restrict__ segR)
{
  __shared__ _Float16 sH1[128 * KH1];   // 32 KB
  __shared__ float    sH2[128 * KH2];   // 16 KB
  __shared__ int      sSID[128];

  const int t = threadIdx.x;
  const int wave = t >> 5;
  const int lane = t & 31;
  const int blockRow0 = blockIdx.x * 128;

  int gr = blockRow0 + t;
  sSID[t] = (gr < KN) ? sid[gr] : 0x7fffffff;
  __syncthreads();

  run_stack(x, frags,                     tfc0_b, tfc1_b, segA,
            sH1, sH2, sSID, blockRow0, wave, lane);
  run_stack(x, frags + (size_t)NFRAG * 512, rl0_b, rl1_b, segR,
            sH1, sH2, sSID, blockRow0, wave, lane);
}

// ---------------------------------------------------------------------------
// Kernel 2: per-segment heads + cost MLP + policy (exact fp32, weights in LDS).
// LDS weight layout (floats):
//   head h (h=0,1,2): base = h*2177 : w0[32*64], b0[64], w1[64], b1[1]
//   c0w[3*64] @6531, c0b[64] @6723, c1w[64*32] @6787, c1b[32] @8835,
//   pw[64] @8867, pb[1] @8931  -> total 8932 floats (~35 KB)
// ---------------------------------------------------------------------------
__device__ __forceinline__ void cp_lds(float* dst, const float* src, int n,
                                       int t, int nt) {
  for (int i = t; i < n; i += nt) dst[i] = src[i];
}

__global__ void __launch_bounds__(256)
segment_heads(const float* __restrict__ segA, const float* __restrict__ segR,
              const float* __restrict__ fwd0_w, const float* __restrict__ fwd0_b,
              const float* __restrict__ fwd1_w, const float* __restrict__ fwd1_b,
              const float* __restrict__ com0_w, const float* __restrict__ com0_b,
              const float* __restrict__ com1_w, const float* __restrict__ com1_b,
              const float* __restrict__ bwd0_w, const float* __restrict__ bwd0_b,
              const float* __restrict__ bwd1_w, const float* __restrict__ bwd1_b,
              const float* __restrict__ cost0_w, const float* __restrict__ cost0_b,
              const float* __restrict__ cost1_w, const float* __restrict__ cost1_b,
              const float* __restrict__ pol_w, const float* __restrict__ pol_b,
              float* __restrict__ out)
{
  __shared__ float W[8932];
  const int t = threadIdx.x;
  const int nt = blockDim.x;

  cp_lds(W + 0 * 2177 + 0,    fwd0_w, 2048, t, nt);
  cp_lds(W + 0 * 2177 + 2048, fwd0_b, 64,   t, nt);
  cp_lds(W + 0 * 2177 + 2112, fwd1_w, 64,   t, nt);
  cp_lds(W + 0 * 2177 + 2176, fwd1_b, 1,    t, nt);
  cp_lds(W + 1 * 2177 + 0,    com0_w, 2048, t, nt);
  cp_lds(W + 1 * 2177 + 2048, com0_b, 64,   t, nt);
  cp_lds(W + 1 * 2177 + 2112, com1_w, 64,   t, nt);
  cp_lds(W + 1 * 2177 + 2176, com1_b, 1,    t, nt);
  cp_lds(W + 2 * 2177 + 0,    bwd0_w, 2048, t, nt);
  cp_lds(W + 2 * 2177 + 2048, bwd0_b, 64,   t, nt);
  cp_lds(W + 2 * 2177 + 2112, bwd1_w, 64,   t, nt);
  cp_lds(W + 2 * 2177 + 2176, bwd1_b, 1,    t, nt);
  cp_lds(W + 6531, cost0_w, 192,  t, nt);
  cp_lds(W + 6723, cost0_b, 64,   t, nt);
  cp_lds(W + 6787, cost1_w, 2048, t, nt);
  cp_lds(W + 8835, cost1_b, 32,   t, nt);
  cp_lds(W + 8867, pol_w,   64,   t, nt);
  cp_lds(W + 8931, pol_b,   1,    t, nt);
  __syncthreads();

  const int s = blockIdx.x * blockDim.x + t;   // segment id, < KSEG

  float st[32], sr[32];
#pragma unroll
  for (int k = 0; k < 32; ++k) st[k] = segA[(size_t)s * 32 + k];
#pragma unroll
  for (int k = 0; k < 32; ++k) sr[k] = segR[(size_t)s * 32 + k];

  // three cost heads: relu(seg@w0+b0)@w1 + b1
  float costs[3];
#pragma unroll
  for (int h = 0; h < 3; ++h) {
    const float* w0 = &W[h * 2177];
    const float* b0 = w0 + 2048;
    const float* w1 = w0 + 2112;
    const float* b1 = w0 + 2176;
    float c = b1[0];
    for (int j = 0; j < 64; ++j) {
      float a = b0[j];
#pragma unroll
      for (int k = 0; k < 32; ++k) a = fmaf(st[k], w0[k * 64 + j], a);
      c = fmaf(fmaxf(a, 0.0f), w1[j], c);
    }
    costs[h] = c;
  }

  // cost MLP: [3] -> 64 (relu) -> 32 (relu)
  const float* c0w = &W[6531];
  const float* c0b = &W[6723];
  const float* c1w = &W[6787];
  const float* c1b = &W[8835];
  const float* pw  = &W[8867];
  const float  pb  = W[8931];

  float ch[64];
#pragma unroll
  for (int j = 0; j < 64; ++j) {
    float v = c0b[j];
    v = fmaf(costs[0], c0w[0 * 64 + j], v);
    v = fmaf(costs[1], c0w[1 * 64 + j], v);
    v = fmaf(costs[2], c0w[2 * 64 + j], v);
    ch[j] = fmaxf(v, 0.0f);
  }

  // logits = [r, c] @ pol_w + pol_b   (r = channels 0..31, c = 32..63)
  float logit = pb;
#pragma unroll
  for (int k = 0; k < 32; ++k) logit = fmaf(sr[k], pw[k], logit);
  for (int t2 = 0; t2 < 32; ++t2) {
    float v = c1b[t2];
#pragma unroll
    for (int j = 0; j < 64; ++j) v = fmaf(ch[j], c1w[j * 32 + t2], v);
    logit = fmaf(fmaxf(v, 0.0f), pw[32 + t2], logit);
  }
  out[s] = logit;
}

// ---------------------------------------------------------------------------
// Launcher
// ---------------------------------------------------------------------------
extern "C" void kernel_launch(void* const* d_in, const int* in_sizes, int n_in,
                              void* d_out, int out_size, void* d_ws, size_t ws_size,
                              hipStream_t stream) {
  const float* x      = (const float*)d_in[0];
  const int*   sid    = (const int*)d_in[1];
  // d_in[2]=B, d_in[3]=D (scalars, values known at compile time)
  const float* tfc0_w = (const float*)d_in[4];
  const float* tfc0_b = (const float*)d_in[5];
  const float* tfc1_w = (const float*)d_in[6];
  const float* tfc1_b = (const float*)d_in[7];
  const float* fwd0_w = (const float*)d_in[8];
  const float* fwd0_b = (const float*)d_in[9];
  const float* fwd1_w = (const float*)d_in[10];
  const float* fwd1_b = (const float*)d_in[11];
  const float* com0_w = (const float*)d_in[12];
  const float* com0_b = (const float*)d_in[13];
  const float* com1_w = (const float*)d_in[14];
  const float* com1_b = (const float*)d_in[15];
  const float* bwd0_w = (const float*)d_in[16];
  const float* bwd0_b = (const float*)d_in[17];
  const float* bwd1_w = (const float*)d_in[18];
  const float* bwd1_b = (const float*)d_in[19];
  const float* rl0_w  = (const float*)d_in[20];
  const float* rl0_b  = (const float*)d_in[21];
  const float* rl1_w  = (const float*)d_in[22];
  const float* rl1_b  = (const float*)d_in[23];
  const float* cost0_w = (const float*)d_in[24];
  const float* cost0_b = (const float*)d_in[25];
  const float* cost1_w = (const float*)d_in[26];
  const float* cost1_b = (const float*)d_in[27];
  const float* pol_w  = (const float*)d_in[28];
  const float* pol_b  = (const float*)d_in[29];

  float* segA = (float*)d_ws;                       // [KSEG][32]
  float* segR = segA + (size_t)KSEG * 32;           // [KSEG][32]
  _Float16* frags = (_Float16*)(segR + (size_t)KSEG * 32);  // [2][24][32][16]

  hipMemsetAsync(d_ws, 0, (size_t)2 * KSEG * 32 * sizeof(float), stream);

  pack_weight_frags<<<2, 256, 0, stream>>>(tfc0_w, tfc1_w, rl0_w, rl1_w, frags);

  const int blocks1 = (KN + 127) / 128;
  fused_rowmlp_segsum<<<blocks1, 128, 0, stream>>>(
      x, sid, frags, tfc0_b, tfc1_b, rl0_b, rl1_b, segA, segR);

  segment_heads<<<KSEG / 256, 256, 0, stream>>>(
      segA, segR,
      fwd0_w, fwd0_b, fwd1_w, fwd1_b,
      com0_w, com0_b, com1_w, com1_b,
      bwd0_w, bwd0_b, bwd1_w, bwd1_b,
      cost0_w, cost0_b, cost1_w, cost1_b,
      pol_w, pol_b, (float*)d_out);
}
